// LSTM_MAE_19533511262663
// MI455X (gfx1250) — compile-verified, matrix operand order checked by reading, not run on hardware
//
#include <hip/hip_runtime.h>
#include <hip/hip_bf16.h>

typedef __attribute__((ext_vector_type(16))) __bf16 v16bf;
typedef __attribute__((ext_vector_type(8)))  float  v8f;

#define E    128
#define T    256
#define BATCH 512
#define NID  16
#define G4   512          // 4*E gate columns
#define XS   257          // x row stride (T+1)
#define ROWS 16           // batch rows per workgroup (one WMMA M tile)
#define NTHREADS 512      // 16 waves
#define MAXTILES 64

// K-index inside a 16-bit 16x32 A (or 32x16 B) WMMA fragment:
// VGPR j (j=e>>1) holds K={2j,2j+1} (lanes 0-15) / K+8 (lanes 16-31),
// VGPRs 4..7 cover K=16..31 with the same pattern.
__device__ __forceinline__ int kmap(int e, int lane) {
    int vg = e >> 1;
    return ((vg < 4) ? 0 : 16) + ((lane >> 4) << 3) + ((vg & 3) << 1) + (e & 1);
}

// branch-free fast tanh: 2*rcp(1+exp(-2x)) - 1   (v_mul,v_exp,v_add,v_rcp,v_fma)
__device__ __forceinline__ float fast_tanh(float x) {
    return 2.0f * __builtin_amdgcn_rcpf(1.0f + __expf(-2.0f * x)) - 1.0f;
}

// Shared recurrent core. Weights W are (512 x 128) row-major f32 in global,
// converted to bf16 B-fragments held in registers for the whole time loop.
// 16 waves; wave w owns gate columns [32w, 32w+32) as 2 WMMA N-tiles.
template<bool DEC>
__device__ void lstm_core(
    const float* __restrict__ W,
    const float* __restrict__ wih,
    const float* __restrict__ bias,
    float*   s_seq,    // [ROWS*T]
    float*   s_gates,  // [ROWS*G4]  (activated gates; i-slot reused as f32 h)
    float*   s_c,      // [ROWS*E]
    __bf16*  s_hb,     // [ROWS*E]
    const float* s_wl, // [E]      (DEC)
    float    bl,       //          (DEC)
    const float* s_u,  // [ROWS]   (DEC)
    const int*  s_row, // [ROWS]   (DEC) global batch row or -1
    int      b0,       // (ENC) first batch row
    float* __restrict__ hEnc, float* __restrict__ cEnc, // (ENC) outputs
    float* __restrict__ out)                            // (DEC) output (B,T)
{
    const int tid  = threadIdx.x;
    const int lane = tid & 31;
    const int wid  = tid >> 5;
    const int nl   = lane & 15;

    // per-wave activation constants (hoisted; no control flow in the loop)
    // waves 0-7: i,f (sigmoid)  waves 8-11: g (tanh)  waves 12-15: o (sigmoid)
    const bool istanh = ((wid >> 2) == 2);
    const float ka = istanh ? -2.0f : -1.0f;
    const float aa = istanh ?  2.0f :  1.0f;
    const float ca = istanh ? -1.0f :  0.0f;

    // ---- Preload loop-invariant B fragments (weights) + per-column params
    v16bf Bf[2][4];
    float c_wih[2], c_b[2];
#pragma unroll
    for (int nt = 0; nt < 2; ++nt) {
        const int g = (wid * 2 + nt) * 16 + nl;   // gate column this lane owns
        c_wih[nt] = wih[g];
        c_b[nt]   = bias[g];
        const float* wr = W + g * E;
#pragma unroll
        for (int ks = 0; ks < 4; ++ks) {
            v16bf f;
#pragma unroll
            for (int e = 0; e < 16; ++e)
                f[e] = (__bf16)wr[ks * 32 + kmap(e, lane)];
            Bf[nt][ks] = f;
        }
    }
    __syncthreads();   // LDS init by caller is now visible

    for (int t = 0; t < T; ++t) {
        // ---- A fragments: h (16x128 bf16) from LDS (merged to ds_load_b128)
        v16bf A[4];
#pragma unroll
        for (int ks = 0; ks < 4; ++ks) {
            v16bf f;
#pragma unroll
            for (int e = 0; e < 16; ++e)
                f[e] = s_hb[nl * E + ks * 32 + kmap(e, lane)];
            A[ks] = f;
        }
        // ---- per-row step inputs (hoisted out of the N-tile loop)
        float utv[8];
#pragma unroll
        for (int r = 0; r < 8; ++r) {
            const int m = r + ((lane >> 4) << 3);
            if (DEC) utv[r] = (t == 0) ? s_u[m] : s_seq[m * T + (T - t)];
            else     utv[r] = s_seq[m * T + t];
        }
        // ---- gates tile: 2 N-tiles x (K=128 as 4 WMMA steps)
#pragma unroll
        for (int nt = 0; nt < 2; ++nt) {
            v8f acc = {};
#pragma unroll
            for (int ks = 0; ks < 4; ++ks)
                acc = __builtin_amdgcn_wmma_f32_16x16x32_bf16(
                    false, A[ks], false, Bf[nt][ks], (short)0, acc, false, false);
            const int g = (wid * 2 + nt) * 16 + nl;
#pragma unroll
            for (int r = 0; r < 8; ++r) {
                const int m = r + ((lane >> 4) << 3);   // C/D row layout
                float v = acc[r] + utv[r] * c_wih[nt] + c_b[nt];
                // act(x) = aa * rcp(1 + exp(ka*x)) + ca   (branch-free)
                v = aa * __builtin_amdgcn_rcpf(1.0f + __expf(ka * v)) + ca;
                s_gates[m * G4 + g] = v;
            }
        }
        __syncthreads();
        // ---- state update: 16*128 elems, 4 per thread
#pragma unroll
        for (int r = 0; r < 4; ++r) {
            const int idx = tid * 4 + r;
            const int m = idx >> 7, e = idx & 127;
            const float iv = s_gates[m * G4 + e];
            const float fv = s_gates[m * G4 + 128 + e];
            const float gv = s_gates[m * G4 + 256 + e];
            const float ov = s_gates[m * G4 + 384 + e];
            const float c  = fv * s_c[m * E + e] + iv * gv;
            const float h  = ov * fast_tanh(c);
            s_c[m * E + e]  = c;
            s_hb[m * E + e] = (__bf16)h;
            s_gates[m * G4 + e] = h;   // keep f32 h (i-slot already consumed)
        }
        __syncthreads();
        if (DEC) {
            // readout: out_t = h . Wl + bl ; 8 lanes per row + butterfly reduce
            if (tid < 128) {
                const int m = tid >> 3, ch = tid & 7;
                float s = 0.f;
#pragma unroll
                for (int j = 0; j < 16; ++j) {
                    const int e = ch * 16 + j;
                    s += s_gates[m * G4 + e] * s_wl[e];
                }
                s += __shfl_xor(s, 1);
                s += __shfl_xor(s, 2);
                s += __shfl_xor(s, 4);
                if (ch == 0) {
                    const int rr = s_row[m];
                    if (rr >= 0) out[rr * T + (T - 1 - t)] = s + bl;
                }
            }
            __syncthreads();
        }
    }
    if (!DEC) {
        // final state writeback (f32 h preserved in s_gates i-slot)
        for (int idx = tid; idx < ROWS * E; idx += NTHREADS) {
            const int m = idx >> 7, e = idx & 127;
            hEnc[(b0 + m) * E + e] = s_gates[m * G4 + e];
            cEnc[(b0 + m) * E + e] = s_c[m * E + e];
        }
    }
}

// ------------------------- prep: group batch rows by id -------------------
__global__ __launch_bounds__(256) void prep_kernel(
    const float* __restrict__ x,
    int* __restrict__ order, int* __restrict__ tileId,
    int* __restrict__ tileRows, int* __restrict__ nTiles)
{
    __shared__ int cnt[NID], cur[NID], rs[NID + 1];
    __shared__ int s_tid[MAXTILES], s_tk[MAXTILES], s_nt;
    const int tid = threadIdx.x;
    if (tid < NID) { cnt[tid] = 0; cur[tid] = 0; }
    __syncthreads();
    for (int bb = tid; bb < BATCH; bb += 256) {
        const int id = (int)x[bb * XS + T];
        atomicAdd(&cnt[id], 1);
    }
    __syncthreads();
    if (tid == 0) {
        rs[0] = 0;
        int nt = 0;
        for (int i = 0; i < NID; ++i) {
            rs[i + 1] = rs[i] + cnt[i];
            const int ti = (cnt[i] + ROWS - 1) / ROWS;
            for (int k = 0; k < ti; ++k) { s_tid[nt] = i; s_tk[nt] = k; ++nt; }
        }
        s_nt = nt;
        *nTiles = nt;
    }
    __syncthreads();
    for (int bb = tid; bb < BATCH; bb += 256) {
        const int id = (int)x[bb * XS + T];
        const int p = atomicAdd(&cur[id], 1);
        order[rs[id] + p] = bb;
    }
    __syncthreads();
    const int nt = s_nt;
    for (int i = tid; i < nt * ROWS; i += 256) {
        const int tt = i / ROWS, j = i % ROWS;
        const int iid = s_tid[tt], k = s_tk[tt];
        const int pos = k * ROWS + j;
        tileRows[i] = (pos < cnt[iid]) ? order[rs[iid] + pos] : -1;
        if (j == 0) tileId[tt] = iid;
    }
}

// ------------------------------ encoder ----------------------------------
__global__ __launch_bounds__(NTHREADS) void enc_kernel(
    const float* __restrict__ x, const float* __restrict__ wih,
    const float* __restrict__ whh, const float* __restrict__ b,
    float* __restrict__ hEnc, float* __restrict__ cEnc)
{
    __shared__ float  s_seq[ROWS * T];
    __shared__ float  s_gates[ROWS * G4];
    __shared__ float  s_c[ROWS * E];
    __shared__ __bf16 s_hb[ROWS * E];
    const int tid = threadIdx.x;
    const int b0 = blockIdx.x * ROWS;
    for (int i = tid; i < ROWS * E; i += NTHREADS) { s_c[i] = 0.f; s_hb[i] = (__bf16)0.f; }
    for (int i = tid; i < ROWS * T; i += NTHREADS) {
        const int m = i / T, t = i % T;
        s_seq[i] = x[(b0 + m) * XS + t];
    }
    lstm_core<false>(whh, wih, b, s_seq, s_gates, s_c, s_hb,
                     nullptr, 0.f, nullptr, nullptr, b0, hEnc, cEnc, nullptr);
}

// ------------------------------ decoder ----------------------------------
__global__ __launch_bounds__(NTHREADS) void dec_kernel(
    const float* __restrict__ x,
    const float* __restrict__ dwih, const float* __restrict__ dwhh,
    const float* __restrict__ db,
    const float* __restrict__ lw, const float* __restrict__ lb,
    const float* __restrict__ hEnc, const float* __restrict__ cEnc,
    const int* __restrict__ tileId, const int* __restrict__ tileRows,
    const int* __restrict__ nTiles,
    float* __restrict__ out)
{
    __shared__ float  s_seq[ROWS * T];
    __shared__ float  s_gates[ROWS * G4];
    __shared__ float  s_c[ROWS * E];
    __shared__ __bf16 s_hb[ROWS * E];
    __shared__ float  s_wl[E];
    __shared__ float  s_u[ROWS];
    __shared__ int    s_row[ROWS];

    if ((int)blockIdx.x >= *nTiles) return;   // uniform per block
    const int tid = threadIdx.x;
    const int id = tileId[blockIdx.x];
    if (tid < ROWS) s_row[tid] = tileRows[blockIdx.x * ROWS + tid];
    if (tid < E)    s_wl[tid] = lw[id * E + tid];
    __syncthreads();
    const float bl = lb[id];
    for (int i = tid; i < ROWS * E; i += NTHREADS) {
        const int m = i >> 7, e = i & 127;
        const int r = s_row[m];
        const float h = (r >= 0) ? hEnc[r * E + e] : 0.f;
        const float c = (r >= 0) ? cEnc[r * E + e] : 0.f;
        s_c[i] = c; s_hb[i] = (__bf16)h;
    }
    for (int i = tid; i < ROWS * T; i += NTHREADS) {
        const int m = i / T, t = i % T;
        const int r = s_row[m];
        s_seq[i] = (r >= 0) ? x[r * XS + t] : 0.f;
    }
    __syncthreads();
    // u0 = h_enc . Wl + bl : 8 lanes per row + butterfly reduce
    if (tid < 128) {
        const int m = tid >> 3, ch = tid & 7;
        const int r = s_row[m];
        float s = 0.f;
        if (r >= 0) {
#pragma unroll
            for (int j = 0; j < 16; ++j) {
                const int e = ch * 16 + j;
                s += hEnc[r * E + e] * s_wl[e];
            }
        }
        s += __shfl_xor(s, 1);
        s += __shfl_xor(s, 2);
        s += __shfl_xor(s, 4);
        if (ch == 0) s_u[m] = s + bl;
    }
    lstm_core<true>(dwhh + (size_t)id * G4 * E, dwih + id * G4, db + id * G4,
                    s_seq, s_gates, s_c, s_hb, s_wl, bl, s_u, s_row,
                    0, nullptr, nullptr, out);
}

// ------------------------------ launcher ---------------------------------
extern "C" void kernel_launch(void* const* d_in, const int* in_sizes, int n_in,
                              void* d_out, int out_size, void* d_ws, size_t ws_size,
                              hipStream_t stream)
{
    const float* x    = (const float*)d_in[0];
    const float* ewih = (const float*)d_in[1];
    const float* ewhh = (const float*)d_in[2];
    const float* eb   = (const float*)d_in[3];
    const float* dwih = (const float*)d_in[4];
    const float* dwhh = (const float*)d_in[5];
    const float* db   = (const float*)d_in[6];
    const float* lw   = (const float*)d_in[7];
    const float* lb   = (const float*)d_in[8];
    float* out = (float*)d_out;

    float* hEnc   = (float*)d_ws;                 // 512*128 f32
    float* cEnc   = hEnc + BATCH * E;             // 512*128 f32
    int*   order  = (int*)(cEnc + BATCH * E);     // 512
    int*   tileId = order + BATCH;                // 64
    int*   tileRows = tileId + MAXTILES;          // 64*16
    int*   nTiles   = tileRows + MAXTILES * ROWS; // 1

    hipLaunchKernelGGL(prep_kernel, dim3(1), dim3(256), 0, stream,
                       x, order, tileId, tileRows, nTiles);
    hipLaunchKernelGGL(enc_kernel, dim3(BATCH / ROWS), dim3(NTHREADS), 0, stream,
                       x, ewih, ewhh, eb, hEnc, cEnc);
    hipLaunchKernelGGL(dec_kernel, dim3(MAXTILES), dim3(NTHREADS), 0, stream,
                       x, dwih, dwhh, db, lw, lb, hEnc, cEnc,
                       tileId, tileRows, nTiles, out);
}
